// WaveNetVocoder_13511967113379
// MI455X (gfx1250) — compile-verified, matrix operand order checked by reading, not run on hardware
//
#include <hip/hip_runtime.h>
#include <hip/hip_bf16.h>
#include <math.h>

// ---------------------------------------------------------------------------
// WaveNet vocoder for MI455X (gfx1250). Channel reductions on
// v_wmma_f32_16x16x32_f16. Activations in (B, T, C) layout so fragment octets
// are contiguous. Tile staging uses GLOBAL_LOAD_ASYNC_TO_LDS_B128 (ASYNCcnt)
// when the toolchain exposes the builtin, else global_load_b128->ds_store_b128.
// Each workgroup covers 64 time steps (4 N-tiles per wave): every A-fragment
// load feeds 4 WMMAs (8 per k-step in dilgate).
// ---------------------------------------------------------------------------

typedef __attribute__((ext_vector_type(16))) _Float16 v16h;
typedef __attribute__((ext_vector_type(8)))  _Float16 v8h;
typedef __attribute__((ext_vector_type(8)))  float    v8f;
typedef __attribute__((ext_vector_type(4)))  int      v4i;

#define T_LEN   8192
#define NBATCH  2
#define RES_C   512
#define SKIP_C  256
#define NLAYERS 30

#if defined(__has_builtin)
#  if __has_builtin(__builtin_amdgcn_global_load_async_to_lds_b128) && \
      __has_builtin(__builtin_amdgcn_s_wait_asynccnt)
#    define USE_ASYNC_LDS 1
#  endif
#endif
#ifndef USE_ASYNC_LDS
#  define USE_ASYNC_LDS 0
#endif

// kk (0..31) inside a 16x32 K-chunk -> (hi, e) per CDNA5 16-bit fragment
// layout (lanes 0-15: K=0..7,16..23; lanes 16-31: K=8..15,24..31).
__device__ __forceinline__ void kk_to_he(int kk, int& hi, int& e) {
  hi = (kk >> 3) & 1;
  e  = (kk & 7) + ((kk >= 16) ? 8 : 0);
}

// ---------------- weight conversion (f32 -> f16, fragment-swizzled) --------

// dil_w: (30, 1024, 512, 3) -> per layer (tap, m_tile[64], k_tile[16], lane[32], e[16])
__global__ __launch_bounds__(256) void cvt_dil(const float* __restrict__ src,
                                               _Float16* __restrict__ dst) {
  int tid = blockIdx.x * 256 + threadIdx.x;
  const int total = NLAYERS * 1024 * 512 * 3;
  if (tid >= total) return;
  int tap   = tid % 3;
  int cin   = (tid / 3) % 512;
  int cout  = (tid / 1536) % 1024;
  int layer = tid / (1536 * 1024);
  int hi, e; kk_to_he(cin & 31, hi, e);
  int lane = hi * 16 + (cout & 15);
  int mt   = cout >> 4;
  int kt   = cin >> 5;
  int off  = layer * (3 * 64 * 16 * 512) + (((tap * 64 + mt) * 16 + kt) * 512) + lane * 16 + e;
  dst[off] = (_Float16)src[tid];
}

// res_w (30,512,512) + skip_w (30,256,512) -> per layer (m_tile[48], k_tile[16], lane, e)
__global__ __launch_bounds__(256) void cvt_rs(const float* __restrict__ res_w,
                                              const float* __restrict__ skip_w,
                                              _Float16* __restrict__ dst) {
  int tid = blockIdx.x * 256 + threadIdx.x;
  const int total = NLAYERS * 768 * 512;
  if (tid >= total) return;
  int cin   = tid % 512;
  int row   = (tid / 512) % 768;
  int layer = tid / (512 * 768);
  float v = (row < 512) ? res_w[(layer * 512 + row) * 512 + cin]
                        : skip_w[(layer * 256 + (row - 512)) * 512 + cin];
  int hi, e; kk_to_he(cin & 31, hi, e);
  int lane = hi * 16 + (row & 15);
  int mt   = row >> 4;
  int kt   = cin >> 5;
  int off  = layer * (768 * 512) + ((mt * 16 + kt) * 512) + lane * 16 + e;
  dst[off] = (_Float16)v;
}

// ---------------- start conv + mel conditioning ((B,T,C) output) -----------

__global__ __launch_bounds__(256) void init_x(const float* __restrict__ mel,
                                              const float* __restrict__ audio,
                                              const float* __restrict__ start_w,
                                              const float* __restrict__ start_b,
                                              const float* __restrict__ mel_w,
                                              const float* __restrict__ mel_b,
                                              _Float16* __restrict__ xp) {
  int tid = blockIdx.x * 256 + threadIdx.x;  // B*T*512, c fastest
  int c = tid & 511;
  int t = (tid >> 9) & (T_LEN - 1);
  int b = tid >> 22;
  float acc = start_w[c] * audio[b * T_LEN + t] + start_b[c] + mel_b[c];
  const float* mrow = mel_w + c * 80;
  const float* mcol = mel + (b * 80) * T_LEN + t;
  #pragma unroll 4
  for (int m = 0; m < 80; ++m) acc += mrow[m] * mcol[m * T_LEN];
  xp[tid] = (_Float16)acc;   // tid == (b*T + t)*512 + c
}

__global__ __launch_bounds__(256) void zero_f32(float* __restrict__ p, int n) {
  int tid = blockIdx.x * 256 + threadIdx.x;
  if (tid < n) p[tid] = 0.f;
}

// ---------------- staging: (B,T,512) f16 tile (64 cols) -> swizzled LDS ----
// LDS layout: [ktile(16)][col(64)][hi(2)][e(16)] halves = 64 KB.
// chunk q (0..4095): oct = q&63 (8 channels), col = q>>6.

__device__ __forceinline__ void stage_tile64(_Float16* lds, const _Float16* __restrict__ src,
                                             int b, int t0, int tshift, int tid) {
  #pragma unroll
  for (int q16 = 0; q16 < 16; ++q16) {
    int q   = tid + (q16 << 8);
    int oct = q & 63;
    int col = q >> 6;
    int t   = t0 + col + tshift;
    int ldsoff = ((oct >> 2) << 11) + (col << 5) + ((oct & 1) << 4) + (((oct >> 1) & 1) << 3);
#if USE_ASYNC_LDS
    if ((unsigned)t < (unsigned)T_LEN) {
      __builtin_amdgcn_global_load_async_to_lds_b128(
          (v4i*)&src[((b * T_LEN + t) << 9) + (oct << 3)],
          (v4i*)&lds[ldsoff],
          0, 0);
    } else {
      v8h z = {};
      *(v8h*)&lds[ldsoff] = z;        // dilation padding
    }
#else
    v8h v = {};
    if ((unsigned)t < (unsigned)T_LEN)
      v = *(const v8h*)&src[((b * T_LEN + t) << 9) + (oct << 3)];
    *(v8h*)&lds[ldsoff] = v;
#endif
  }
#if USE_ASYNC_LDS
  __builtin_amdgcn_s_wait_asynccnt(0);
#endif
}

// fast gated activation: tanh(f)*sigmoid(g) with v_exp_f32 only
__device__ __forceinline__ float gate_act(float f, float g) {
  float th = 1.f - 2.f / (__expf(2.f * f) + 1.f);
  float sg = 1.f / (1.f + __expf(-g));
  return th * sg;
}

// ---------------- dilated conv + gated activation (WMMA) -------------------
// grid: (T/64, 512/128, B), 256 threads = 8 waves; wave owns 16 gate-channels
// x 64 time steps (4 N-tiles). f uses h-rows c, g uses h-rows c+512.

__global__ __launch_bounds__(256) void dilgate(const _Float16* __restrict__ x,
                                               const _Float16* __restrict__ wdil,  // layer, swizzled
                                               const float* __restrict__ bdil,    // layer, 1024
                                               _Float16* __restrict__ a_out,
                                               int dil) {
  __shared__ __align__(16) _Float16 lds[32768];  // 64 KB, one tap tile
  const int tid  = threadIdx.x;
  const int lane = tid & 31;
  const int wave = tid >> 5;
  const int t0    = blockIdx.x << 6;
  const int arow0 = (blockIdx.y << 7) + (wave << 4);
  const int b     = blockIdx.z;

  v8f cf[4] = {};
  v8f cg[4] = {};
  const int mtf = arow0 >> 4;      // f m-tile (0..31)
  const int mtg = mtf + 32;        // g m-tile (32..63)
  const int bco = ((lane & 15) << 5) + ((lane >> 4) << 4);   // B frag base (N-tile 0)

  for (int tap = 0; tap < 3; ++tap) {
    __syncthreads();               // previous tap fully consumed
    stage_tile64(lds, x, b, t0, (tap - 1) * dil, tid);
    __syncthreads();

    const _Float16* wtf = wdil + (((tap * 64 + mtf) * 16) << 9) + (lane << 4);
    const _Float16* wtg = wdil + (((tap * 64 + mtg) * 16) << 9) + (lane << 4);
    #pragma unroll 4
    for (int kt = 0; kt < 16; ++kt) {
      v16h af = *(const v16h*)(wtf + (kt << 9));
      v16h ag = *(const v16h*)(wtg + (kt << 9));
      if (kt + 1 < 16) {
        __builtin_prefetch(wtf + ((kt + 1) << 9), 0, 3);
        __builtin_prefetch(wtg + ((kt + 1) << 9), 0, 3);
      }
      #pragma unroll
      for (int j = 0; j < 4; ++j) {
        v16h bj = *(const v16h*)&lds[(kt << 11) + bco + (j << 9)];
        cf[j] = __builtin_amdgcn_wmma_f32_16x16x32_f16(false, af, false, bj, (short)0, cf[j], false, false);
        cg[j] = __builtin_amdgcn_wmma_f32_16x16x32_f16(false, ag, false, bj, (short)0, cg[j], false, false);
      }
    }
  }

  // gated activation + vector store (C layout: elem e -> M = e + 8*(lane>=16))
  const int nloc  = lane & 15;
  const int mbase = (lane >> 4) << 3;
  const int crow  = arow0 + mbase;
  const v8f bf = *(const v8f*)&bdil[crow];
  const v8f bg = *(const v8f*)&bdil[crow + 512];
  #pragma unroll
  for (int j = 0; j < 4; ++j) {
    int t = t0 + (j << 4) + nloc;
    v8h ov;
    #pragma unroll
    for (int e = 0; e < 8; ++e)
      ov[e] = (_Float16)gate_act(cf[j][e] + bf[e], cg[j][e] + bg[e]);
    *(v8h*)&a_out[((b * T_LEN + t) << 9) + crow] = ov;
  }
}

// ---------------- fused residual (512) + skip (256) 1x1 convs (WMMA) -------
// grid: (T/64, 768/128, B). rows 0..511 -> x_out = x_in + res; 512..767 -> skip +=
// Each wave's 16 rows are entirely res or entirely skip (uniform branch).

__global__ __launch_bounds__(256) void resskip(const _Float16* __restrict__ a,
                                               const _Float16* __restrict__ xin,
                                               const _Float16* __restrict__ wrs,  // layer, swizzled
                                               const float* __restrict__ res_b,   // 512
                                               const float* __restrict__ skip_b,  // 256
                                               _Float16* __restrict__ xout,
                                               float* __restrict__ skip) {
  __shared__ __align__(16) _Float16 lds[32768];
  const int tid  = threadIdx.x;
  const int lane = tid & 31;
  const int wave = tid >> 5;
  const int t0   = blockIdx.x << 6;
  const int row0 = (blockIdx.y << 7) + (wave << 4);
  const int b    = blockIdx.z;
  const int bco  = ((lane & 15) << 5) + ((lane >> 4) << 4);

  stage_tile64(lds, a, b, t0, 0, tid);
  __syncthreads();

  v8f acc[4] = {};
  const _Float16* w = wrs + (((row0 >> 4) * 16) << 9) + (lane << 4);
  #pragma unroll 4
  for (int kt = 0; kt < 16; ++kt) {
    v16h af = *(const v16h*)(w + (kt << 9));
    if (kt + 1 < 16) __builtin_prefetch(w + ((kt + 1) << 9), 0, 3);
    #pragma unroll
    for (int j = 0; j < 4; ++j) {
      v16h bj = *(const v16h*)&lds[(kt << 11) + bco + (j << 9)];
      acc[j] = __builtin_amdgcn_wmma_f32_16x16x32_f16(false, af, false, bj, (short)0, acc[j], false, false);
    }
  }

  const int nloc  = lane & 15;
  const int mbase = (lane >> 4) << 3;
  if (row0 < 512) {
    const int crow = row0 + mbase;
    const v8f rb = *(const v8f*)&res_b[crow];
    #pragma unroll
    for (int j = 0; j < 4; ++j) {
      int t = t0 + (j << 4) + nloc;
      int off = ((b * T_LEN + t) << 9) + crow;
      v8h xv = *(const v8h*)&xin[off];
      v8h ov;
      #pragma unroll
      for (int e = 0; e < 8; ++e) ov[e] = (_Float16)((float)xv[e] + acc[j][e] + rb[e]);
      *(v8h*)&xout[off] = ov;
    }
  } else {
    const int sr0 = row0 - 512 + mbase;
    const v8f sb = *(const v8f*)&skip_b[sr0];
    #pragma unroll
    for (int j = 0; j < 4; ++j) {
      int t = t0 + (j << 4) + nloc;
      float* sp = &skip[((b * T_LEN + t) << 8) + sr0];
      v8f s = *(v8f*)sp;
      #pragma unroll
      for (int e = 0; e < 8; ++e) s[e] += acc[j][e] + sb[e];
      *(v8f*)sp = s;
    }
  }
}

// ---------------- head: relu(fc1) then tanh(fc2), (B,T,C) layout -----------

__global__ __launch_bounds__(256) void fc1_k(const float* __restrict__ skip,
                                             const float* __restrict__ w,
                                             const float* __restrict__ bias,
                                             float* __restrict__ y1) {
  int tid = blockIdx.x * 256 + threadIdx.x;  // B*T*256, c fastest
  int c = tid & 255;
  float acc = bias[c];
  const float* wr = w + c * 256;
  const float* sc = skip + (tid - c);        // row base: (b*T+t)*256
  #pragma unroll 4
  for (int k = 0; k < 256; ++k) acc += wr[k] * sc[k];
  y1[tid] = fmaxf(acc, 0.f);
}

__global__ __launch_bounds__(256) void fc2_k(const float* __restrict__ y1,
                                             const float* __restrict__ w,
                                             const float* __restrict__ bias,
                                             float* __restrict__ out) {
  int tid = blockIdx.x * 256 + threadIdx.x;  // B*T
  if (tid >= NBATCH * T_LEN) return;
  float acc = bias[0];
  const float* yc = y1 + (tid << 8);
  #pragma unroll 4
  for (int k = 0; k < 256; ++k) acc += w[k] * yc[k];
  out[tid] = tanhf(acc);                     // (B,1,T) flat == b*T+t
}

// ---------------------------------------------------------------------------

extern "C" void kernel_launch(void* const* d_in, const int* in_sizes, int n_in,
                              void* d_out, int out_size, void* d_ws, size_t ws_size,
                              hipStream_t stream) {
  const float* mel     = (const float*)d_in[0];
  const float* audio   = (const float*)d_in[1];
  const float* start_w = (const float*)d_in[2];
  const float* start_b = (const float*)d_in[3];
  const float* mel_w   = (const float*)d_in[4];
  const float* mel_b   = (const float*)d_in[5];
  const float* dil_w   = (const float*)d_in[6];
  const float* dil_b   = (const float*)d_in[7];
  const float* res_w   = (const float*)d_in[8];
  const float* res_b   = (const float*)d_in[9];
  const float* skip_w  = (const float*)d_in[10];
  const float* skip_b  = (const float*)d_in[11];
  const float* fc1_w   = (const float*)d_in[12];
  const float* fc1_b   = (const float*)d_in[13];
  const float* fc2_w   = (const float*)d_in[14];
  const float* fc2_b   = (const float*)d_in[15];
  float* out = (float*)d_out;

  char* ws = (char*)d_ws;
  // workspace carve-up (all 256B aligned)
  _Float16* wdil = (_Float16*)(ws);                       // 94,371,840 B
  _Float16* wrs  = (_Float16*)(ws + 94371840);            // 23,592,960 B
  _Float16* xp   = (_Float16*)(ws + 117964800);           // 16,777,216 B (B,T,512) f16
  _Float16* xq   = (_Float16*)(ws + 134742016);           // 16,777,216 B
  _Float16* abuf = (_Float16*)(ws + 151519232);           // 16,777,216 B (reused as y1 f32)
  float*    skip = (float*)   (ws + 168296448);           // 16,777,216 B (B,T,256) f32
  float*    y1   = (float*)abuf;

  // 1. weight conversion (swizzle into WMMA fragment layout)
  cvt_dil<<<(NLAYERS * 1024 * 512 * 3 + 255) / 256, 256, 0, stream>>>(dil_w, wdil);
  cvt_rs <<<(NLAYERS * 768 * 512 + 255) / 256, 256, 0, stream>>>(res_w, skip_w, wrs);

  // 2. start conv + mel conditioning -> x (f16), zero skip accumulator
  init_x<<<(NBATCH * 512 * T_LEN) / 256, 256, 0, stream>>>(mel, audio, start_w, start_b,
                                                           mel_w, mel_b, xp);
  zero_f32<<<(NBATCH * 256 * T_LEN) / 256, 256, 0, stream>>>(skip, NBATCH * 256 * T_LEN);

  // 3. 30 layers, ping-pong x
  for (int i = 0; i < NLAYERS; ++i) {
    int d = 1 << (i % 10);
    const _Float16* xs = (i & 1) ? xq : xp;
    _Float16*       xd = (i & 1) ? xp : xq;
    dilgate<<<dim3(T_LEN / 64, 4, NBATCH), 256, 0, stream>>>(
        xs, wdil + (size_t)i * (3 * 64 * 16 * 512), dil_b + i * 1024, abuf, d);
    resskip<<<dim3(T_LEN / 64, 6, NBATCH), 256, 0, stream>>>(
        abuf, xs, wrs + (size_t)i * (768 * 512), res_b + i * 512, skip_b + i * 256, xd, skip);
  }

  // 4. head
  fc1_k<<<(NBATCH * 256 * T_LEN) / 256, 256, 0, stream>>>(skip, fc1_w, fc1_b, y1);
  fc2_k<<<(NBATCH * T_LEN + 255) / 256, 256, 0, stream>>>(y1, fc2_w, fc2_b, out);

  (void)in_sizes; (void)n_in; (void)out_size; (void)ws_size;
}
